// VAREncoderCIFAR_4990751998228
// MI455X (gfx1250) — matrix operand order repaired and checked by reading, not compiled, not run on hardware
//
#include <hip/hip_runtime.h>
#include <hip/hip_bf16.h>
#include <cmath>

// ---------------------------------------------------------------------------
// MI455X / gfx1250, wave32. bf16 WMMA (v_wmma_f32_16x16x32_bf16) everywhere.
// Register-blocked GEMM: one wave owns a 32x64 output macro-tile.
// ---------------------------------------------------------------------------

typedef __attribute__((ext_vector_type(16))) __bf16 v16bf;
typedef __attribute__((ext_vector_type(8)))  __bf16 v8bf;
typedef __attribute__((ext_vector_type(8)))  float  v8f;

constexpr int kBatch = 1024;
constexpr int kTok   = 85;            // N tokens
constexpr int kRows  = kBatch * kTok; // 87040, divisible by 32
constexpr int kDim   = 512;
constexpr int kHeads = 8;
constexpr int kHd    = 64;
constexpr int kMlp   = 2048;
constexpr int kDepth = 6;
constexpr int kPad   = 96;            // 85 padded to 6x16 tiles

__device__ __forceinline__ v16bf cat16(v8bf lo, v8bf hi) {
  return __builtin_shufflevector(lo, hi, 0,1,2,3,4,5,6,7,8,9,10,11,12,13,14,15);
}
__device__ __forceinline__ v8f wmma_bf16(v16bf a, v16bf b, v8f c) {
  // emits v_wmma_f32_16x16x32_bf16
  return __builtin_amdgcn_wmma_f32_16x16x32_bf16(false, a, false, b, (short)0, c, false, false);
}

// ---------------------------------------------------------------------------
// Weight pack: fp32 row-major (K x N) -> bf16 WMMA B-fragment tile layout.
// Tile (kt, nt) of 32x16 stored as 512 contiguous bf16: lane*16 + e,
// with k = kt*32 + e + (lane>=16 ? 16 : 0), n = nt*16 + (lane&15).
// blockIdx.y = layer.
// ---------------------------------------------------------------------------
__global__ __launch_bounds__(256) void pack_w(const float* __restrict__ W,
                                              __bf16* __restrict__ out,
                                              int K, int Nn) {
  const size_t per = (size_t)K * Nn;
  const size_t idx = (size_t)blockIdx.x * 256 + threadIdx.x;
  if (idx >= per) return;
  const float* Wl = W + (size_t)blockIdx.y * per;
  __bf16* ol = out + (size_t)blockIdx.y * per;
  const int Ntl = Nn >> 4;
  const size_t tile = idx >> 9;
  const int within = (int)(idx & 511);
  const int lane = within >> 4, e = within & 15;
  const int kt = (int)(tile / Ntl), nt = (int)(tile % Ntl);
  const int k = kt * 32 + e + ((lane >= 16) ? 16 : 0);
  const int n = nt * 16 + (lane & 15);
  ol[idx] = (__bf16)Wl[(size_t)k * Nn + n];
}

// ---------------------------------------------------------------------------
// LayerNorm over 512 cols, one wave per row, bf16 output.
// ---------------------------------------------------------------------------
__global__ __launch_bounds__(256) void ln_bf16(const float* __restrict__ x,
                                               const float* __restrict__ g,
                                               const float* __restrict__ b,
                                               __bf16* __restrict__ out) {
  const int row  = blockIdx.x * 8 + (threadIdx.x >> 5);
  const int lane = threadIdx.x & 31;
  const float* xr = x + (size_t)row * kDim;
  float v[16];
  float s = 0.f;
#pragma unroll
  for (int i = 0; i < 16; ++i) { v[i] = xr[lane + i * 32]; s += v[i]; }
#pragma unroll
  for (int o = 16; o; o >>= 1) s += __shfl_xor(s, o, 32);
  const float mu = s * (1.0f / 512.0f);
  float vs = 0.f;
#pragma unroll
  for (int i = 0; i < 16; ++i) { float d = v[i] - mu; vs += d * d; }
#pragma unroll
  for (int o = 16; o; o >>= 1) vs += __shfl_xor(vs, o, 32);
  const float inv = rsqrtf(vs * (1.0f / 512.0f) + 1e-5f);
  __bf16* orow = out + (size_t)row * kDim;
#pragma unroll
  for (int i = 0; i < 16; ++i) {
    const int c = lane + i * 32;
    orow[c] = (__bf16)((v[i] - mu) * inv * g[c] + b[c]);
  }
}

// ---------------------------------------------------------------------------
// Register-blocked WMMA GEMM: C[M,N] = A[M,K](bf16 row-major) x Bpacked + bias.
// One wave computes a 32x64 macro-tile: 2 A-frags x 4 B-frags -> 8 wmma per
// 32-deep K step (12 b128 loads per 8 wmma).
// MODE 0: +bias -> bf16 out          (QKV)
// MODE 1: +bias, GELU -> bf16 out    (FC1)
// MODE 2: +bias, +residual -> f32    (proj / FC2, in-place residual ok)
// ---------------------------------------------------------------------------
template<int K, int MODE>
__global__ __launch_bounds__(256) void gemm_wmma(
    const __bf16* __restrict__ A, const __bf16* __restrict__ Bp,
    const float* __restrict__ bias, const float* __restrict__ resid,
    float* __restrict__ outF, __bf16* __restrict__ outB,
    int Mt2, int Nt4) {
  const int wid = blockIdx.x * 8 + (threadIdx.x >> 5);
  if (wid >= Mt2 * Nt4) return;
  const int mt = wid / Nt4, nt = wid % Nt4;
  const int lane = threadIdx.x & 31;
  const int half = lane >> 4, l15 = lane & 15;
  const int Ntiles = Nt4 * 4;                 // 16-wide tiles across N
  const size_t bstep = (size_t)Ntiles * 512;  // elements per K-step of tiles

  const __bf16* Arow0 = A + (size_t)(mt * 32 + l15) * K + half * 16;
  const __bf16* Arow1 = Arow0 + (size_t)16 * K;
  const __bf16* bbase = Bp + (size_t)(nt * 4) * 512 + (size_t)lane * 16;

  v8f acc[2][4] = {};
#pragma unroll 2
  for (int kt = 0; kt < K / 32; ++kt) {
    const __bf16* ap0 = Arow0 + kt * 32;
    const __bf16* ap1 = Arow1 + kt * 32;
    v16bf a0 = cat16(*(const v8bf*)ap0, *(const v8bf*)(ap0 + 8));
    v16bf a1 = cat16(*(const v8bf*)ap1, *(const v8bf*)(ap1 + 8));
    const __bf16* bp = bbase + (size_t)kt * bstep;
    v16bf b[4];
#pragma unroll
    for (int ns = 0; ns < 4; ++ns) b[ns] = *(const v16bf*)(bp + ns * 512);
    // stream-prefetch packed weights a few K-steps ahead (global_prefetch_b8)
    __builtin_prefetch(bp + 4 * bstep, 0, 1);
#pragma unroll
    for (int ns = 0; ns < 4; ++ns) {
      acc[0][ns] = wmma_bf16(a0, b[ns], acc[0][ns]);
      acc[1][ns] = wmma_bf16(a1, b[ns], acc[1][ns]);
    }
  }

  const int Ncols = Ntiles * 16;
#pragma unroll
  for (int ns = 0; ns < 4; ++ns) {
    const int n = nt * 64 + ns * 16 + l15;
    const float bv = bias[n];
#pragma unroll
    for (int ms = 0; ms < 2; ++ms) {
#pragma unroll
      for (int r = 0; r < 8; ++r) {
        const int m = mt * 32 + ms * 16 + r + half * 8;
        const size_t idx = (size_t)m * Ncols + n;
        float v = acc[ms][ns][r] + bv;
        if (MODE == 1) v = 0.5f * v * (1.0f + erff(v * 0.70710678f));
        if (MODE == 2) outF[idx] = v + resid[idx];
        else           outB[idx] = (__bf16)v;
      }
    }
  }
}

// ---------------------------------------------------------------------------
// Attention mask from TOKENS=[1,4,16,64]: cum=[0,1,5,21,85].
// ---------------------------------------------------------------------------
__device__ __forceinline__ int seg_of(int t) {
  return (t == 0) ? 0 : (t < 5 ? 1 : (t < 21 ? 2 : 3));
}
__device__ __forceinline__ bool mask_ok(int q, int k) {
  if (q >= kTok || k >= kTok) return false;
  if (q == 0) return (k == 0) || (k >= 21);
  return (k == 0) || (seg_of(k) == seg_of(q));
}

// ---------------------------------------------------------------------------
// Attention: one block (192 threads = 6 waves) per (batch, head).
// RoPE on Q,K load, S = Q K^T via WMMA, masked softmax, O = P V via WMMA.
// ---------------------------------------------------------------------------
__global__ __launch_bounds__(192) void attn_wmma(const __bf16* __restrict__ qkv,
                                                 const float* __restrict__ coords,
                                                 __bf16* __restrict__ outb) {
  __shared__ __bf16 Qs[kPad][kHd];     // 12 KB
  __shared__ __bf16 Ks[kPad][kHd];     // 12 KB
  __shared__ __bf16 Vt[kHd][kPad];     // 12 KB (transposed: [d][tok])
  __shared__ __bf16 Ps[kPad][kPad];    // 18 KB

  const int bidx = blockIdx.x >> 3;
  const int h    = blockIdx.x & 7;
  const __bf16* base = qkv + (size_t)bidx * kTok * 1536 + h * 64;
  const int tid = threadIdx.x;

  // --- load Q,K with RoPE (token 0 = CLS, no rotation), pad rows 85..95 ---
  for (int it = tid; it < kPad * 16; it += 192) {
    const int t = it >> 4, j = it & 15;
    float q0 = 0, q1 = 0, q2 = 0, q3 = 0, k0 = 0, k1 = 0, k2 = 0, k3 = 0;
    if (t < kTok) {
      const __bf16* qp = base + (size_t)t * 1536;
      const __bf16* kp = qp + 512;
      q0 = (float)qp[j];      q1 = (float)qp[16 + j];
      q2 = (float)qp[32 + j]; q3 = (float)qp[48 + j];
      k0 = (float)kp[j];      k1 = (float)kp[16 + j];
      k2 = (float)kp[32 + j]; k3 = (float)kp[48 + j];
      if (t > 0) {
        const float fr = expf((float)j * -0.14391156858f);   // 10^(-j/16)
        const float ax = coords[(t - 1) * 2 + 0] * fr * 6.28318530718f;
        const float ay = coords[(t - 1) * 2 + 1] * fr * 6.28318530718f;
        const float cx = cosf(ax), sx = sinf(ax);
        const float cy = cosf(ay), sy = sinf(ay);
        float r0 = q0 * cx - q1 * sx, r1 = q0 * sx + q1 * cx;
        float r2 = q2 * cy - q3 * sy, r3 = q2 * sy + q3 * cy;
        q0 = r0; q1 = r1; q2 = r2; q3 = r3;
        r0 = k0 * cx - k1 * sx; r1 = k0 * sx + k1 * cx;
        r2 = k2 * cy - k3 * sy; r3 = k2 * sy + k3 * cy;
        k0 = r0; k1 = r1; k2 = r2; k3 = r3;
      }
    }
    Qs[t][j] = (__bf16)q0; Qs[t][16 + j] = (__bf16)q1;
    Qs[t][32 + j] = (__bf16)q2; Qs[t][48 + j] = (__bf16)q3;
    Ks[t][j] = (__bf16)k0; Ks[t][16 + j] = (__bf16)k1;
    Ks[t][32 + j] = (__bf16)k2; Ks[t][48 + j] = (__bf16)k3;
  }
  // --- load V transposed ---
  for (int it = tid; it < kPad * kHd; it += 192) {
    const int t = it / kHd, d = it % kHd;
    float v = 0.f;
    if (t < kTok) v = (float)base[(size_t)t * 1536 + 1024 + d];
    Vt[d][t] = (__bf16)v;
  }
  __syncthreads();

  const int qt = tid >> 5;                 // wave -> query row tile (0..5)
  const int lane = tid & 31;
  const int half = lane >> 4, l15 = lane & 15;

  // --- S = Q K^T (6 tiles of 16x16, K-dim 64 = 2 wmma) ---
  v8f s[6];
#pragma unroll
  for (int kt = 0; kt < 6; ++kt) {
    v8f acc = {};
#pragma unroll
    for (int kk = 0; kk < 2; ++kk) {
      const __bf16* qrow = &Qs[qt * 16 + l15][kk * 32 + half * 16];
      v16bf a = cat16(*(const v8bf*)qrow, *(const v8bf*)(qrow + 8));
      const __bf16* krow = &Ks[kt * 16 + l15][kk * 32 + half * 16];
      v16bf b = *(const v16bf*)krow;
      acc = wmma_bf16(a, b, acc);
    }
    s[kt] = acc;
  }
  // --- scale + mask ---
#pragma unroll
  for (int kt = 0; kt < 6; ++kt)
#pragma unroll
    for (int r = 0; r < 8; ++r) {
      const int q = qt * 16 + r + half * 8;
      const int k = kt * 16 + l15;
      s[kt][r] = mask_ok(q, k) ? s[kt][r] * 0.125f : -1e30f;
    }
  // --- softmax over 96 cols: C layout = lane holds one col per fragment ---
  v8f mx = s[0];
#pragma unroll
  for (int kt = 1; kt < 6; ++kt)
#pragma unroll
    for (int r = 0; r < 8; ++r) mx[r] = fmaxf(mx[r], s[kt][r]);
#pragma unroll
  for (int r = 0; r < 8; ++r)
#pragma unroll
    for (int o = 8; o; o >>= 1) mx[r] = fmaxf(mx[r], __shfl_xor(mx[r], o, 32));
  v8f sum = {};
#pragma unroll
  for (int kt = 0; kt < 6; ++kt)
#pragma unroll
    for (int r = 0; r < 8; ++r) { s[kt][r] = expf(s[kt][r] - mx[r]); sum[r] += s[kt][r]; }
#pragma unroll
  for (int r = 0; r < 8; ++r)
#pragma unroll
    for (int o = 8; o; o >>= 1) sum[r] += __shfl_xor(sum[r], o, 32);
  // --- write P to LDS in row-major so it can be reloaded as A fragments ---
#pragma unroll
  for (int r = 0; r < 8; ++r) {
    const float inv = 1.0f / sum[r];
#pragma unroll
    for (int kt = 0; kt < 6; ++kt)
      Ps[qt * 16 + r + half * 8][kt * 16 + l15] = (__bf16)(s[kt][r] * inv);
  }
  __syncthreads();

  // --- O = P V : 4 col tiles of 16, K-dim 96 = 3 wmma ---
#pragma unroll
  for (int nt = 0; nt < 4; ++nt) {
    v8f o = {};
#pragma unroll
    for (int kt = 0; kt < 3; ++kt) {
      const __bf16* prow = &Ps[qt * 16 + l15][kt * 32 + half * 16];
      v16bf a = cat16(*(const v8bf*)prow, *(const v8bf*)(prow + 8));
      const __bf16* vrow = &Vt[nt * 16 + l15][kt * 32 + half * 16];
      v16bf b = *(const v16bf*)vrow;
      o = wmma_bf16(a, b, o);
    }
#pragma unroll
    for (int r = 0; r < 8; ++r) {
      const int t = qt * 16 + r + half * 8;
      if (t < kTok) {
        const int d = nt * 16 + l15;
        outb[((size_t)bidx * kTok + t) * kDim + h * 64 + d] = (__bf16)o[r];
      }
    }
  }
}

// ---------------------------------------------------------------------------
// Host launcher
// ---------------------------------------------------------------------------
extern "C" void kernel_launch(void* const* d_in, const int* in_sizes, int n_in,
                              void* d_out, int out_size, void* d_ws, size_t ws_size,
                              hipStream_t stream) {
  (void)in_sizes; (void)n_in; (void)out_size; (void)ws_size;
  const float* x_in   = (const float*)d_in[0];
  const float* coords = (const float*)d_in[1];
  const float* ln1_g  = (const float*)d_in[2];
  const float* ln1_b  = (const float*)d_in[3];
  const float* qkv_w  = (const float*)d_in[4];
  const float* qkv_b  = (const float*)d_in[5];
  const float* proj_w = (const float*)d_in[6];
  const float* proj_b = (const float*)d_in[7];
  const float* ln2_g  = (const float*)d_in[8];
  const float* ln2_b  = (const float*)d_in[9];
  const float* fc1_w  = (const float*)d_in[10];
  const float* fc1_b  = (const float*)d_in[11];
  const float* fc2_w  = (const float*)d_in[12];
  const float* fc2_b  = (const float*)d_in[13];
  float* x = (float*)d_out;   // activations live in d_out (fp32, in-place residuals)

  // workspace layout (bytes, all 256-aligned)
  char* ws = (char*)d_ws;
  constexpr size_t kHbufB = (size_t)kRows * kDim * 2;    //  89,128,960
  constexpr size_t kBigB  = (size_t)kRows * kMlp * 2;    // 356,515,840 (qkv reuses this)
  __bf16* hbuf = (__bf16*)ws;                            // LN out / attn out
  __bf16* big  = (__bf16*)(ws + kHbufB);                 // qkv (bf16) then mlp hidden
  __bf16* wq   = (__bf16*)(ws + kHbufB + kBigB);         // packed weights
  __bf16* wp   = wq  + (size_t)kDepth * kDim * (3 * kDim);
  __bf16* wf1  = wp  + (size_t)kDepth * kDim * kDim;
  __bf16* wf2  = wf1 + (size_t)kDepth * kDim * kMlp;

  // x <- input
  hipMemcpyAsync(x, x_in, (size_t)kRows * kDim * sizeof(float),
                 hipMemcpyDeviceToDevice, stream);

  // pack all weights to bf16 WMMA fragment layout (once per call)
  pack_w<<<dim3((kDim * 3 * kDim) / 256, kDepth), 256, 0, stream>>>(qkv_w, wq, kDim, 3 * kDim);
  pack_w<<<dim3((kDim * kDim) / 256, kDepth), 256, 0, stream>>>(proj_w, wp, kDim, kDim);
  pack_w<<<dim3((kDim * kMlp) / 256, kDepth), 256, 0, stream>>>(fc1_w, wf1, kDim, kMlp);
  pack_w<<<dim3((kMlp * kDim) / 256, kDepth), 256, 0, stream>>>(fc2_w, wf2, kMlp, kDim);

  const int Mt2 = kRows / 32;                      // 2720 macro-row tiles
  const int lnBlocks = kRows / 8;                  // 10880
  for (int l = 0; l < kDepth; ++l) {
    // LN1 -> bf16
    ln_bf16<<<lnBlocks, 256, 0, stream>>>(x, ln1_g + l * kDim, ln1_b + l * kDim, hbuf);
    // QKV: (87040x512) x (512x1536) + b -> bf16   (N64 tiles = 24)
    gemm_wmma<512, 0><<<(Mt2 * 24) / 8, 256, 0, stream>>>(
        hbuf, wq + (size_t)l * kDim * 3 * kDim, qkv_b + l * 3 * kDim,
        nullptr, nullptr, big, Mt2, 24);
    // attention (RoPE + masked softmax + PV), out bf16 row-major (87040x512)
    attn_wmma<<<kBatch * kHeads, 192, 0, stream>>>(big, coords, hbuf);
    // proj + residual -> fp32 x (in-place)        (N64 tiles = 8)
    gemm_wmma<512, 2><<<(Mt2 * 8) / 8, 256, 0, stream>>>(
        hbuf, wp + (size_t)l * kDim * kDim, proj_b + l * kDim,
        x, x, nullptr, Mt2, 8);
    // LN2 -> bf16
    ln_bf16<<<lnBlocks, 256, 0, stream>>>(x, ln2_g + l * kDim, ln2_b + l * kDim, hbuf);
    // FC1 + GELU -> bf16 (87040x2048)             (N64 tiles = 32)
    gemm_wmma<512, 1><<<(Mt2 * 32) / 8, 256, 0, stream>>>(
        hbuf, wf1 + (size_t)l * kDim * kMlp, fc1_b + l * kMlp,
        nullptr, nullptr, big, Mt2, 32);
    // FC2 + residual -> fp32 x (in-place)         (N64 tiles = 8)
    gemm_wmma<2048, 2><<<(Mt2 * 8) / 8, 256, 0, stream>>>(
        big, wf2 + (size_t)l * kMlp * kDim, fc2_b + l * kDim,
        x, x, nullptr, Mt2, 8);
  }
}
